// ElmanTanh_7206955123453
// MI455X (gfx1250) — matrix operand order repaired and checked by reading, not compiled, not run on hardware
//
#include <hip/hip_runtime.h>
#include <hip/hip_bf16.h>

typedef __attribute__((ext_vector_type(16))) __bf16   v16bf;
typedef __attribute__((ext_vector_type(8)))  float    v8f;
typedef __attribute__((ext_vector_type(4)))  unsigned u32x4;

#define T_STEPS 512
#define BATCH   64
#define DIN     1024
#define HID     1024
#define TWOH    2048

// ws layout (bytes)
#define XBF_OFF  0u            // [32768][1024] bf16 = 64 MB
#define WXBF_OFF 67108864u     // [2048][1024]  bf16 = 4 MB
#define RBF_OFF  71303168u     // [2048][1024]  bf16 = 4 MB
#define HBF_OFF  75497472u     // 2 x [64][1024] bf16 = 256 KB
#define CNT_OFF  75759616u     // barrier counter

__device__ __forceinline__ unsigned short f32_bf16(float f) {
  unsigned u = __float_as_uint(f);
  u += 0x7FFFu + ((u >> 16) & 1u);           // round-to-nearest-even
  return (unsigned short)(u >> 16);
}
__device__ __forceinline__ float bf16_f32(unsigned short h) {
  return __uint_as_float(((unsigned)h) << 16);
}

union FragBF { v16bf v; u32x4 q[2]; };

// ---------------------------------------------------------------------------
// Kernel 1: convert x, Wx, R, h0 -> bf16 workspace; reset barrier counter
// ---------------------------------------------------------------------------
__global__ void conv_kernel(const float* __restrict__ x, const float* __restrict__ h0,
                            const float* __restrict__ Wx, const float* __restrict__ R,
                            unsigned short* __restrict__ xbf, unsigned short* __restrict__ wxbf,
                            unsigned short* __restrict__ rbf, unsigned short* __restrict__ hbf,
                            unsigned* cnt) {
  long i = (long)blockIdx.x * blockDim.x + threadIdx.x;
  const long NX = (long)T_STEPS * BATCH * DIN;   // 33,554,432
  const long NW = (long)TWOH * DIN;              // 2,097,152
  if (i < NX) xbf[i] = f32_bf16(x[i]);
  if (i < NW) { wxbf[i] = f32_bf16(Wx[i]); rbf[i] = f32_bf16(R[i]); }
  if (i < BATCH * HID)                    hbf[i] = f32_bf16(h0[i]);   // buffer 0 = h0
  else if (i < 2 * BATCH * HID)           hbf[i] = 0;                 // buffer 1
  if (i == 0) *cnt = 0u;
}

// ---------------------------------------------------------------------------
// Kernel 2: xproj = x @ Wx.T + bias -> bf16, staged inside d_out.
// Register-blocked 32x64 macro-tile per wave (2 M-frags x 4 N-frags):
// 8 v_wmma per K-step for 6 fragment loads -> 4x A / 2x B traffic reduction.
// ---------------------------------------------------------------------------
__global__ void xproj_kernel(const unsigned short* __restrict__ xbf,
                             const unsigned short* __restrict__ wxbf,
                             const float* __restrict__ bias,
                             unsigned short* __restrict__ xp) {
  const int lane = threadIdx.x & 31;
  const int wave = threadIdx.x >> 5;
  const int unit = blockIdx.x * 8 + wave;        // 0..32767
  const int m0 = (unit >> 5) * 32;               // 1024 M-macrotiles over T*B
  const int n0 = (unit & 31) * 64;               // 32 N-macrotiles over 2H

  const int akb = (lane < 16) ? 0 : 16;          // contiguous K-split (matches B)
  const int bkb = (lane < 16) ? 0 : 16;          // contiguous K-split (matches A)
  const unsigned short* pa0 = xbf + (long)(m0 + (lane & 15)) * DIN + akb;
  const unsigned short* pa1 = pa0 + 16 * DIN;
  const unsigned short* pb0 = wxbf + (long)(n0 +  0 + (lane & 15)) * DIN + bkb;
  const unsigned short* pb1 = pb0 + 16 * DIN;
  const unsigned short* pb2 = pb0 + 32 * DIN;
  const unsigned short* pb3 = pb0 + 48 * DIN;

  v8f acc[2][4] = {};
  for (int k0 = 0; k0 < DIN; k0 += 32) {
    FragBF a0, a1, b0, b1, b2, b3;
    a0.q[0] = *(const u32x4*)(pa0 + k0); a0.q[1] = *(const u32x4*)(pa0 + k0 + 8);
    a1.q[0] = *(const u32x4*)(pa1 + k0); a1.q[1] = *(const u32x4*)(pa1 + k0 + 8);
    b0.q[0] = *(const u32x4*)(pb0 + k0); b0.q[1] = *(const u32x4*)(pb0 + k0 + 8);
    b1.q[0] = *(const u32x4*)(pb1 + k0); b1.q[1] = *(const u32x4*)(pb1 + k0 + 8);
    b2.q[0] = *(const u32x4*)(pb2 + k0); b2.q[1] = *(const u32x4*)(pb2 + k0 + 8);
    b3.q[0] = *(const u32x4*)(pb3 + k0); b3.q[1] = *(const u32x4*)(pb3 + k0 + 8);
    acc[0][0] = __builtin_amdgcn_wmma_f32_16x16x32_bf16(false, a0.v, false, b0.v, (short)0, acc[0][0], false, false);
    acc[0][1] = __builtin_amdgcn_wmma_f32_16x16x32_bf16(false, a0.v, false, b1.v, (short)0, acc[0][1], false, false);
    acc[0][2] = __builtin_amdgcn_wmma_f32_16x16x32_bf16(false, a0.v, false, b2.v, (short)0, acc[0][2], false, false);
    acc[0][3] = __builtin_amdgcn_wmma_f32_16x16x32_bf16(false, a0.v, false, b3.v, (short)0, acc[0][3], false, false);
    acc[1][0] = __builtin_amdgcn_wmma_f32_16x16x32_bf16(false, a1.v, false, b0.v, (short)0, acc[1][0], false, false);
    acc[1][1] = __builtin_amdgcn_wmma_f32_16x16x32_bf16(false, a1.v, false, b1.v, (short)0, acc[1][1], false, false);
    acc[1][2] = __builtin_amdgcn_wmma_f32_16x16x32_bf16(false, a1.v, false, b2.v, (short)0, acc[1][2], false, false);
    acc[1][3] = __builtin_amdgcn_wmma_f32_16x16x32_bf16(false, a1.v, false, b3.v, (short)0, acc[1][3], false, false);
  }

#pragma unroll
  for (int j = 0; j < 4; ++j) {
    const int ocol = n0 + 16 * j + (lane & 15);
    const float bv = bias[ocol];
#pragma unroll
    for (int i = 0; i < 2; ++i) {
      const int orow0 = m0 + 16 * i + 8 * (lane >> 4);
#pragma unroll
      for (int v = 0; v < 8; ++v)
        xp[(long)(orow0 + v) * TWOH + ocol] = f32_bf16(acc[i][j][v] + bv);
    }
  }
}

// ---------------------------------------------------------------------------
// Kernel 3: persistent recurrent scan. 32 blocks x 8 waves = 256 waves:
// wave (m,p) owns cand tile (m, p) and gate tile (m, p+64) -> local tanh*tanh,
// one grid barrier per timestep. xp[t+1] prefetched before the barrier.
// ---------------------------------------------------------------------------
__global__ void scan_kernel(const unsigned short* __restrict__ rbf,
                            unsigned short* __restrict__ hbf,
                            float* __restrict__ out, unsigned* cnt) {
  const int lane = threadIdx.x & 31;
  const int wave = threadIdx.x >> 5;
  const int wid  = blockIdx.x * 8 + wave;   // 0..255
  const int m = wid >> 6;                   // 0..3   batch row tile
  const int p = wid & 63;                   // 0..63  hidden col tile

  unsigned short* xpAll = (unsigned short*)out;   // bf16 xproj staged in d_out

  const int arow = m * 16 + (lane & 15);
  const int akb  = (lane < 16) ? 0 : 16;
  const int bkb  = (lane < 16) ? 0 : 16;
  const unsigned short* pbc = rbf + (long)(p * 16 + (lane & 15)) * HID + bkb;        // cand rows of R
  const unsigned short* pbg = rbf + (long)(HID + p * 16 + (lane & 15)) * HID + bkb;  // gate rows of R

  const int crow0 = m * 16 + 8 * (lane >> 4);   // batch row of acc element v=0
  const int ccol  = p * 16 + (lane & 15);       // hidden column

  // prefetch xp[0]
  float xpc[8], xpg[8];
#pragma unroll
  for (int v = 0; v < 8; ++v) {
    long base = (long)(crow0 + v) * TWOH;
    xpc[v] = bf16_f32(xpAll[base + ccol]);
    xpg[v] = bf16_f32(xpAll[base + HID + ccol]);
  }

  unsigned short* hcur = hbf;                 // holds h_{t-1} (starts as h0)
  unsigned short* hnxt = hbf + BATCH * HID;

  for (int t = 0; t < T_STEPS; ++t) {
    v8f accc, accg;
#pragma unroll
    for (int v = 0; v < 8; ++v) { accc[v] = xpc[v]; accg[v] = xpg[v]; }

    const unsigned short* pa = hcur + (long)arow * HID + akb;
    for (int k0 = 0; k0 < HID; k0 += 32) {
      FragBF a, bc, bg;
      a.q[0]  = *(const u32x4*)(pa + k0);
      a.q[1]  = *(const u32x4*)(pa + k0 + 8);
      bc.q[0] = *(const u32x4*)(pbc + k0);
      bc.q[1] = *(const u32x4*)(pbc + k0 + 8);
      bg.q[0] = *(const u32x4*)(pbg + k0);
      bg.q[1] = *(const u32x4*)(pbg + k0 + 8);
      accc = __builtin_amdgcn_wmma_f32_16x16x32_bf16(false, a.v, false, bc.v,
                                                     (short)0, accc, false, false);
      accg = __builtin_amdgcn_wmma_f32_16x16x32_bf16(false, a.v, false, bg.v,
                                                     (short)0, accg, false, false);
    }

    float hn[8];
#pragma unroll
    for (int v = 0; v < 8; ++v) hn[v] = tanhf(accc[v]) * tanhf(accg[v]);

    // prefetch xp[t+1] (region untouched until after the barrier below)
    if (t + 1 < T_STEPS) {
#pragma unroll
      for (int v = 0; v < 8; ++v) {
        long base = ((long)(t + 1) * BATCH + (crow0 + v)) * TWOH;
        xpc[v] = bf16_f32(xpAll[base + ccol]);
        xpg[v] = bf16_f32(xpAll[base + HID + ccol]);
      }
    }

    // write hs[t] (f32, overwrites consumed xp[t] bytes) and h_t (bf16)
#pragma unroll
    for (int v = 0; v < 8; ++v) {
      out[((long)t * BATCH + (crow0 + v)) * HID + ccol] = hn[v];
      hnxt[(crow0 + v) * HID + ccol] = f32_bf16(hn[v]);
    }

    // CDNA5 split counters: drain this wave's loads (prefetch) and stores
    asm volatile("s_wait_loadcnt 0x0" ::: "memory");
    asm volatile("s_wait_storecnt 0x0" ::: "memory");
    __threadfence();
    __syncthreads();
    if (threadIdx.x == 0) {
      __hip_atomic_fetch_add(cnt, 1u, __ATOMIC_ACQ_REL, __HIP_MEMORY_SCOPE_AGENT);
      const unsigned target = (unsigned)(t + 1) * gridDim.x;
      while (__hip_atomic_load(cnt, __ATOMIC_ACQUIRE, __HIP_MEMORY_SCOPE_AGENT) < target)
        __builtin_amdgcn_s_sleep(1);
    }
    __syncthreads();

    unsigned short* tmp = hcur; hcur = hnxt; hnxt = tmp;
  }
}

// ---------------------------------------------------------------------------
extern "C" void kernel_launch(void* const* d_in, const int* in_sizes, int n_in,
                              void* d_out, int out_size, void* d_ws, size_t ws_size,
                              hipStream_t stream) {
  const float* x    = (const float*)d_in[0];
  const float* h0   = (const float*)d_in[1];
  const float* Wx   = (const float*)d_in[2];
  const float* R    = (const float*)d_in[3];
  const float* bias = (const float*)d_in[4];

  char* ws = (char*)d_ws;
  unsigned short* xbf  = (unsigned short*)(ws + XBF_OFF);
  unsigned short* wxbf = (unsigned short*)(ws + WXBF_OFF);
  unsigned short* rbf  = (unsigned short*)(ws + RBF_OFF);
  unsigned short* hbf  = (unsigned short*)(ws + HBF_OFF);
  unsigned*       cnt  = (unsigned*)(ws + CNT_OFF);

  // 1) fp32 -> bf16 conversion + state init (memory bound, ~200 MB traffic)
  conv_kernel<<<131072, 256, 0, stream>>>(x, h0, Wx, R, xbf, wxbf, rbf, hbf, cnt);
  // 2) fully-parallel input projection GEMM, bf16 result staged in d_out
  xproj_kernel<<<4096, 256, 0, stream>>>(xbf, wxbf, bias, (unsigned short*)d_out);
  // 3) persistent recurrent scan with per-step grid barrier
  scan_kernel<<<32, 256, 0, stream>>>(rbf, hbf, (float*)d_out, cnt);
}
